// VQVAE_45140106281398
// MI455X (gfx1250) — compile-verified, matrix operand order checked by reading, not run on hardware
//
#include <hip/hip_runtime.h>
#include <stdint.h>

typedef __attribute__((ext_vector_type(16))) _Float16 v16h;
typedef __attribute__((ext_vector_type(8)))  _Float16 v8h;
typedef __attribute__((ext_vector_type(8)))  float    v8f;

#define BN_EPS 1e-5f

// ---------------------------------------------------------------------------
// Elementwise / packing kernels
// ---------------------------------------------------------------------------

__global__ void k_f32_to_f16(const float* __restrict__ x, _Float16* __restrict__ y, int n) {
  int i = blockIdx.x * 256 + threadIdx.x;
  if (i < n) y[i] = (_Float16)x[i];
}

// dct1_w: (Cin, Cout, 4, 4) -> Wt[oc][ic*16 + r*4 + s]  (f16)
__global__ void k_pack_convT_w(const float* __restrict__ w, _Float16* __restrict__ wt,
                               int Cin, int Cout) {
  int i = blockIdx.x * 256 + threadIdx.x;
  int total = Cin * Cout * 16;
  if (i >= total) return;
  int k  = i % (Cin * 16);
  int oc = i / (Cin * 16);
  int ic = k >> 4;
  int rs = k & 15;
  wt[i] = (_Float16)w[((ic * Cout + oc) << 4) + rs];
}

// Per-channel batch stats over (N,H,W): scale = g*rsqrt(var+eps), shift = b - mean*scale
__global__ void k_bn_stats(const float* __restrict__ x, const float* __restrict__ g,
                           const float* __restrict__ b, float* __restrict__ scale,
                           float* __restrict__ shift, int C, int HW, int Nb) {
  __shared__ float s_sum[256];
  __shared__ float s_sq[256];
  int c = blockIdx.x;
  int count = Nb * HW;
  float sum = 0.f, sq = 0.f;
  for (int idx = threadIdx.x; idx < count; idx += 256) {
    int n = idx / HW, j = idx - n * HW;
    float v = x[(size_t)(n * C + c) * HW + j];
    sum += v; sq += v * v;
  }
  s_sum[threadIdx.x] = sum; s_sq[threadIdx.x] = sq;
  __syncthreads();
  for (int o = 128; o > 0; o >>= 1) {
    if ((int)threadIdx.x < o) {
      s_sum[threadIdx.x] += s_sum[threadIdx.x + o];
      s_sq[threadIdx.x]  += s_sq[threadIdx.x + o];
    }
    __syncthreads();
  }
  if (threadIdx.x == 0) {
    float m = s_sum[0] / (float)count;
    float v = s_sq[0] / (float)count - m * m;
    float sc = g[c] * rsqrtf(v + BN_EPS);
    scale[c] = sc;
    shift[c] = b[c] - m * sc;
  }
}

// y = f16(relu(x*scale[c] + shift[c])), NCHW ; HW passed as log2
__global__ void k_bn_relu_f16(const float* __restrict__ x, const float* __restrict__ scale,
                              const float* __restrict__ shift, _Float16* __restrict__ y,
                              int total, int logHW) {
  int i = blockIdx.x * 256 + threadIdx.x;
  if (i >= total) return;
  int c = (i >> logHW) & 255;
  float v = fmaxf(x[i] * scale[c] + shift[c], 0.f);
  y[i] = (_Float16)v;
}

// z_e (NCHW f32, 16x256x32x32) -> Aq[m][c] f16 with m = (n*32+h)*32+w
__global__ void k_pack_ze(const float* __restrict__ ze, _Float16* __restrict__ aq) {
  int i = blockIdx.x * 256 + threadIdx.x;
  if (i >= 16 * 256 * 32 * 32) return;
  int w = i & 31, h = (i >> 5) & 31, c = (i >> 10) & 255, n = i >> 18;
  int m = (n << 10) + (h << 5) + w;
  aq[(size_t)m * 256 + c] = (_Float16)ze[i];
}

// cnorm[k] = sum_d codebook[k][d]^2   (512 blocks x 256 threads)
__global__ void k_cnorm(const float* __restrict__ cb, float* __restrict__ cn) {
  __shared__ float s[256];
  float v = cb[(size_t)blockIdx.x * 256 + threadIdx.x];
  s[threadIdx.x] = v * v;
  __syncthreads();
  for (int o = 128; o > 0; o >>= 1) {
    if ((int)threadIdx.x < o) s[threadIdx.x] += s[threadIdx.x + o];
    __syncthreads();
  }
  if (threadIdx.x == 0) cn[blockIdx.x] = s[0];
}

// ---------------------------------------------------------------------------
// Direct fp32 convs for the C=3 boundary layers (tiny FLOPs, skip WMMA)
// ---------------------------------------------------------------------------

// ec1: x(16,3,128,128) * w(256,3,4,4) s2 p1 + b -> y(16,256,64,64)
__global__ void k_conv_ec1(const float* __restrict__ x, const float* __restrict__ w,
                           const float* __restrict__ b, float* __restrict__ y) {
  int i = blockIdx.x * 256 + threadIdx.x;
  if (i >= 16 * 256 * 64 * 64) return;
  int wo = i & 63, ho = (i >> 6) & 63, oc = (i >> 12) & 255, n = i >> 20;
  float acc = b[oc];
  for (int c = 0; c < 3; ++c)
    for (int r = 0; r < 4; ++r) {
      int hi = 2 * ho - 1 + r;
      if (hi < 0 || hi >= 128) continue;
      for (int s = 0; s < 4; ++s) {
        int wi = 2 * wo - 1 + s;
        if (wi < 0 || wi >= 128) continue;
        acc += x[((n * 3 + c) * 128 + hi) * 128 + wi] * w[((oc * 3 + c) * 4 + r) * 4 + s];
      }
    }
  y[i] = acc;
}

// dct2: ConvT(256->3, k4 s2 p1) on f16 act(16,256,64,64) -> x_rec(16,3,128,128)
__global__ void k_convT_dct2(const _Float16* __restrict__ x, const float* __restrict__ w,
                             const float* __restrict__ b, float* __restrict__ y) {
  int i = blockIdx.x * 256 + threadIdx.x;
  if (i >= 16 * 3 * 128 * 128) return;
  int wo = i & 127, ho = (i >> 7) & 127;
  int oc = (i >> 14) % 3, n = i / (3 * 16384);
  float acc = b[oc];
  for (int r = 0; r < 4; ++r) {
    int a2 = ho + 1 - r;
    if (a2 < 0 || (a2 & 1)) continue;
    int hi = a2 >> 1;
    if (hi >= 64) continue;
    for (int s = 0; s < 4; ++s) {
      int b2 = wo + 1 - s;
      if (b2 < 0 || (b2 & 1)) continue;
      int wi = b2 >> 1;
      if (wi >= 64) continue;
      const _Float16* xp = &x[((size_t)(n * 256) * 64 + hi) * 64 + wi];
      const float* wp = &w[(oc * 4 + r) * 4 + s];
      float a = 0.f;
      for (int ic = 0; ic < 256; ++ic)
        a += (float)xp[(size_t)ic * 4096] * wp[(size_t)ic * 48];
      acc += a;
    }
  }
  y[i] = acc;
}

// ---------------------------------------------------------------------------
// WMMA implicit-GEMM: out[M,N] = alpha * (A @ B^T) + bias + residual
//   A: implicit im2col over f16 NCHW (C=256) activation, or flat [M,K]
//   B: packed f16 weights [N][K] row-major, K multiple of 32, N multiple of 64
// Block tile 128x64, 8 waves, each 32x32 via 2x2 WMMA f32_16x16x32_f16 tiles.
// Geometry is compile-time (shifts/masks/const-div). Gather loads are
// UNCONDITIONAL with clamped addresses + post-load zero select, so the 16
// element loads batch under a single s_wait_loadcnt instead of serializing.
// B tile is staged with gfx1250 async global->LDS (ASYNCcnt) b128 copies.
// ---------------------------------------------------------------------------

#define MODE_CONV  0
#define MODE_FLAT  1
#define MODE_CONVT 2

template <int MODE, int R, int S, int STRIDE, int PAD,
          int LOGH, int LOGW, int LOGHO, int LOGWO>
__launch_bounds__(256)
__global__ void k_gemm_wmma(const _Float16* __restrict__ A, const _Float16* __restrict__ Bw,
                            const float* __restrict__ bias, const float* __restrict__ res,
                            float* __restrict__ out,
                            int M, int N, int K, float alpha, int out_rowmajor) {
  __shared__ __align__(16) _Float16 As[128][40];  // 80B rows: 16B-aligned chunks
  __shared__ __align__(16) _Float16 Bs[64][40];

  const int tid  = threadIdx.x;
  const int lane = tid & 31;
  const int wave = tid >> 5;
  const int m0 = blockIdx.x * 128;
  const int n0 = blockIdx.y * 64;
  const int wm = (wave >> 1) * 32;
  const int wn = (wave & 1) * 32;
  constexpr int RS = R * S;
  constexpr int Hd = 1 << LOGH, Wd = 1 << LOGW;

  // ---- per-thread A-staging geometry (one LDS row per thread), hoisted ----
  const int arow = tid >> 1;           // 0..127
  const int acol = (tid & 1) << 4;     // 0 or 16
  const int am_raw = m0 + arow;
  const bool am_ok = am_raw < M;
  const int am = am_ok ? am_raw : (M - 1);   // clamped, loads always safe
  int a_h0 = 0, a_w0 = 0;
  const _Float16* ap = A;
  if (MODE != MODE_FLAT) {
    int wo   = am & ((1 << LOGWO) - 1);
    int ho   = (am >> LOGWO) & ((1 << LOGHO) - 1);
    int nimg = am >> (LOGWO + LOGHO);
    if (MODE == MODE_CONV) { a_h0 = ho * STRIDE - PAD; a_w0 = wo * STRIDE - PAD; }
    else                   { a_h0 = ho + 1;            a_w0 = wo + 1; }  // ConvT k4 s2 p1
    ap = A + ((size_t)(nimg * 256) << (LOGH + LOGW));
  }

  // ---- per-thread B-staging geometry: one b128 async copy per lane ----
  const int brow = tid >> 2;           // 0..63
  const int bch  = (tid & 3) << 3;     // f16 offset of this 16B chunk
  const _Float16* bsrc = &Bw[(size_t)(n0 + brow) * K + bch];
  const uint32_t  bdst = (uint32_t)(uintptr_t)&Bs[brow][bch];

  v8f acc[2][2] = {};

  for (int k0 = 0; k0 < K; k0 += 32) {
    // B tile: async global -> LDS, 64x32 f16 = 256 lanes x 16B  (ASYNCcnt)
    asm volatile("global_load_async_to_lds_b128 %0, %1, off"
                 :: "v"(bdst), "v"(bsrc + k0) : "memory");
    if (k0 + 32 < K)  // prefetch next weight K-tile (global_prefetch_b8)
      __builtin_prefetch(bsrc + k0 + 32, 0, 1);

    // A tile: implicit-im2col gather, batched unconditional loads
    {
      v8h v0 = {}, v1 = {};
      if (MODE == MODE_FLAT) {
        if (am_ok) {
          v0 = *(const v8h*)&A[(size_t)am * K + k0 + acol];
          v1 = *(const v8h*)&A[(size_t)am * K + k0 + acol + 8];
        }
      } else {
        _Float16 raw[16];
        bool okm[16];
        // phase 1: issue all 16 loads (clamped addresses, no predication)
#pragma unroll
        for (int j = 0; j < 16; ++j) {
          int k = k0 + acol + j;
          int c = k / RS;
          int rs = k - c * RS;
          int r = rs / S;
          int s = rs - r * S;
          int hi, wi;
          bool ok;
          if (MODE == MODE_CONV) {
            hi = a_h0 + r; wi = a_w0 + s;
            ok = (hi >= 0) & (hi < Hd) & (wi >= 0) & (wi < Wd);
          } else {
            int a2 = a_h0 - r, b2 = a_w0 - s;
            hi = a2 >> 1; wi = b2 >> 1;
            ok = (a2 >= 0) & (b2 >= 0) & ((a2 & 1) == 0) & ((b2 & 1) == 0) &
                 (hi < Hd) & (wi < Wd);
          }
          int hic = min(max(hi, 0), Hd - 1);
          int wic = min(max(wi, 0), Wd - 1);
          raw[j] = ap[((size_t)c << (LOGH + LOGW)) + (hic << LOGW) + wic];
          okm[j] = ok & am_ok;
        }
        // phase 2: zero-pad select + pack
#pragma unroll
        for (int j = 0; j < 16; ++j) {
          _Float16 v = okm[j] ? raw[j] : (_Float16)0.f;
          if (j < 8) v0[j] = v; else v1[j - 8] = v;
        }
      }
      *(v8h*)&As[arow][acol]     = v0;
      *(v8h*)&As[arow][acol + 8] = v1;
    }

    asm volatile("s_wait_asynccnt 0x0" ::: "memory");  // own async B writes visible
    __syncthreads();                                   // + ds waits, all waves arrived

    // fragments per documented 16-bit A/B VGPR layouts
    const int khalf = lane >> 4;
    v16h af[2], bf[2];
#pragma unroll
    for (int ti = 0; ti < 2; ++ti) {
      int row = wm + 16 * ti + (lane & 15);
      v8h lo = *(const v8h*)&As[row][khalf * 8];
      v8h hi = *(const v8h*)&As[row][16 + khalf * 8];
      af[ti] = __builtin_shufflevector(lo, hi, 0, 1, 2, 3, 4, 5, 6, 7,
                                       8, 9, 10, 11, 12, 13, 14, 15);
    }
#pragma unroll
    for (int tj = 0; tj < 2; ++tj) {
      int row = wn + 16 * tj + (lane & 15);
      v8h lo = *(const v8h*)&Bs[row][khalf * 8];
      v8h hi = *(const v8h*)&Bs[row][16 + khalf * 8];
      bf[tj] = __builtin_shufflevector(lo, hi, 0, 1, 2, 3, 4, 5, 6, 7,
                                       8, 9, 10, 11, 12, 13, 14, 15);
    }
#pragma unroll
    for (int ti = 0; ti < 2; ++ti)
#pragma unroll
      for (int tj = 0; tj < 2; ++tj)
        acc[ti][tj] = __builtin_amdgcn_wmma_f32_16x16x32_f16(
            false, af[ti], false, bf[tj], (short)0, acc[ti][tj], false, false);
    __syncthreads();
  }

  // ---- epilogue: alpha, bias, residual, scatter (shift/mask decode) ----
#pragma unroll
  for (int ti = 0; ti < 2; ++ti)
    for (int tj = 0; tj < 2; ++tj)
#pragma unroll
      for (int v = 0; v < 8; ++v) {
        int m = m0 + wm + 16 * ti + ((lane >> 4) * 8 + v);
        int n = n0 + wn + 16 * tj + (lane & 15);
        if (m < M) {
          float val = alpha * acc[ti][tj][v];
          if (bias) val += bias[n];
          size_t oidx;
          if (out_rowmajor) {
            oidx = (size_t)m * N + n;
          } else {
            int wo   = m & ((1 << LOGWO) - 1);
            int ho   = (m >> LOGWO) & ((1 << LOGHO) - 1);
            int nimg = m >> (LOGWO + LOGHO);
            oidx = (((size_t)(nimg * N + n)) << (LOGHO + LOGWO)) + (ho << LOGWO) + wo;
          }
          if (res) val += res[oidx];
          out[oidx] = val;
        }
      }
}

// ---------------------------------------------------------------------------
// VQ argmin + codebook gather: one block per latent vector m
// ---------------------------------------------------------------------------
__global__ void k_argmin_zq(const float* __restrict__ scores, const float* __restrict__ cb,
                            float* __restrict__ zq) {
  __shared__ float sv[256];
  __shared__ int   si[256];
  __shared__ int   qsh;
  int m = blockIdx.x;
  int tid = threadIdx.x;
  float best = scores[(size_t)m * 512 + tid];
  int bi = tid;
  float v2 = scores[(size_t)m * 512 + tid + 256];
  if (v2 < best) { best = v2; bi = tid + 256; }
  sv[tid] = best; si[tid] = bi;
  __syncthreads();
  for (int o = 128; o > 0; o >>= 1) {
    if (tid < o) {
      if (sv[tid + o] < sv[tid] || (sv[tid + o] == sv[tid] && si[tid + o] < si[tid])) {
        sv[tid] = sv[tid + o]; si[tid] = si[tid + o];
      }
    }
    __syncthreads();
  }
  if (tid == 0) qsh = si[0];
  __syncthreads();
  int q = qsh;
  int w = m & 31, h = (m >> 5) & 31, n = m >> 10;
  zq[((size_t)(n * 256 + tid) * 32 + h) * 32 + w] = cb[(size_t)q * 256 + tid];
}

// ---------------------------------------------------------------------------
// Host orchestration
// ---------------------------------------------------------------------------

extern "C" void kernel_launch(void* const* d_in, const int* in_sizes, int n_in,
                              void* d_out, int out_size, void* d_ws, size_t ws_size,
                              hipStream_t stream) {
  (void)in_sizes; (void)n_in; (void)out_size; (void)ws_size;

  const float* x        = (const float*)d_in[0];
  const float* ec1_w    = (const float*)d_in[1];
  const float* ec1_b    = (const float*)d_in[2];
  const float* ebn1_g   = (const float*)d_in[3];
  const float* ebn1_b   = (const float*)d_in[4];
  const float* ec2_w    = (const float*)d_in[5];
  const float* ec2_b    = (const float*)d_in[6];
  const float* rb_bn1_g = (const float*)d_in[7];
  const float* rb_bn1_b = (const float*)d_in[8];
  const float* rb_c3_w  = (const float*)d_in[9];
  const float* rb_c3_b  = (const float*)d_in[10];
  const float* rb_bn2_g = (const float*)d_in[11];
  const float* rb_bn2_b = (const float*)d_in[12];
  const float* rb_c1_w  = (const float*)d_in[13];
  const float* rb_c1_b  = (const float*)d_in[14];
  const float* dbn1_g   = (const float*)d_in[15];
  const float* dbn1_b   = (const float*)d_in[16];
  const float* dct1_w   = (const float*)d_in[17];
  const float* dct1_b   = (const float*)d_in[18];
  const float* dbn2_g   = (const float*)d_in[19];
  const float* dbn2_b   = (const float*)d_in[20];
  const float* dct2_w   = (const float*)d_in[21];
  const float* dct2_b   = (const float*)d_in[22];
  const float* codebook = (const float*)d_in[23];

  const int E64 = 16 * 256 * 64 * 64;
  const int E32 = 16 * 256 * 32 * 32;
  const int M32 = 16 * 32 * 32;

  float* out  = (float*)d_out;
  float* xrec = out;
  float* ze   = out + 16 * 3 * 128 * 128;
  float* zq   = ze + E32;

  char* ws = (char*)d_ws;
  size_t off = 0;
  auto carve = [&](size_t bytes) -> char* {
    char* p = ws + off;
    off += (bytes + 255) & ~(size_t)255;
    return p;
  };
  float*    A64  = (float*)carve((size_t)E64 * 4);
  _Float16* X64  = (_Float16*)carve((size_t)E64 * 2);
  float*    P    = (float*)carve((size_t)E32 * 4);
  float*    Q    = (float*)carve((size_t)E32 * 4);
  float*    T    = (float*)carve((size_t)E32 * 4);
  _Float16* X32  = (_Float16*)carve((size_t)E32 * 2);
  float*    SCOR = (float*)carve((size_t)M32 * 512 * 4);
  _Float16* WEC2 = (_Float16*)carve((size_t)256 * 4096 * 2);
  _Float16* WC3  = (_Float16*)carve((size_t)4 * 256 * 2304 * 2);
  _Float16* WC1  = (_Float16*)carve((size_t)4 * 256 * 256 * 2);
  _Float16* WDT1 = (_Float16*)carve((size_t)256 * 4096 * 2);
  _Float16* WCB  = (_Float16*)carve((size_t)512 * 256 * 2);
  float*    SC   = (float*)carve(256 * 4);
  float*    SH   = (float*)carve(256 * 4);
  float*    CN   = (float*)carve(512 * 4);

  auto grid1 = [](int n) { return dim3((n + 255) / 256); };

  // ---- weight packing (conv OIHW layout is already [O][K]) ----
  k_f32_to_f16<<<grid1(256 * 4096), 256, 0, stream>>>(ec2_w, WEC2, 256 * 4096);
  k_f32_to_f16<<<grid1(4 * 256 * 2304), 256, 0, stream>>>(rb_c3_w, WC3, 4 * 256 * 2304);
  k_f32_to_f16<<<grid1(4 * 256 * 256), 256, 0, stream>>>(rb_c1_w, WC1, 4 * 256 * 256);
  k_f32_to_f16<<<grid1(512 * 256), 256, 0, stream>>>(codebook, WCB, 512 * 256);
  k_pack_convT_w<<<grid1(256 * 256 * 16), 256, 0, stream>>>(dct1_w, WDT1, 256, 256);

  // ---- encoder ----
  k_conv_ec1<<<grid1(E64), 256, 0, stream>>>(x, ec1_w, ec1_b, A64);
  k_bn_stats<<<256, 256, 0, stream>>>(A64, ebn1_g, ebn1_b, SC, SH, 256, 4096, 16);
  k_bn_relu_f16<<<grid1(E64), 256, 0, stream>>>(A64, SC, SH, X64, E64, 12);

  // ec2: conv4x4 s2 p1, (256,64,64) -> (256,32,32): M=16384 N=256 K=4096
  k_gemm_wmma<MODE_CONV, 4, 4, 2, 1, 6, 6, 5, 5>
      <<<dim3(128, 4), 256, 0, stream>>>(X64, WEC2, ec2_b, nullptr, P,
                                         M32, 256, 4096, 1.f, 0);

  // ---- residual blocks (pre-act) ----
  auto resblock = [&](int i, const float* in, float* outb) {
    k_bn_stats<<<256, 256, 0, stream>>>(in, rb_bn1_g + i * 256, rb_bn1_b + i * 256,
                                        SC, SH, 256, 1024, 16);
    k_bn_relu_f16<<<grid1(E32), 256, 0, stream>>>(in, SC, SH, X32, E32, 10);
    k_gemm_wmma<MODE_CONV, 3, 3, 1, 1, 5, 5, 5, 5>
        <<<dim3(128, 4), 256, 0, stream>>>(X32, WC3 + (size_t)i * 256 * 2304,
                                           rb_c3_b + i * 256, nullptr, T,
                                           M32, 256, 2304, 1.f, 0);
    k_bn_stats<<<256, 256, 0, stream>>>(T, rb_bn2_g + i * 256, rb_bn2_b + i * 256,
                                        SC, SH, 256, 1024, 16);
    k_bn_relu_f16<<<grid1(E32), 256, 0, stream>>>(T, SC, SH, X32, E32, 10);
    k_gemm_wmma<MODE_CONV, 1, 1, 1, 0, 5, 5, 5, 5>
        <<<dim3(128, 4), 256, 0, stream>>>(X32, WC1 + (size_t)i * 256 * 256,
                                           rb_c1_b + i * 256, in, outb,
                                           M32, 256, 256, 1.f, 0);
  };

  resblock(0, P, Q);
  resblock(1, Q, ze);  // z_e -> d_out

  // ---- vector quantizer: scores = ||c_k||^2 - 2 z.c ; argmin ; gather ----
  k_pack_ze<<<grid1(E32), 256, 0, stream>>>(ze, X32);
  k_cnorm<<<512, 256, 0, stream>>>(codebook, CN);
  k_gemm_wmma<MODE_FLAT, 1, 1, 1, 0, 0, 0, 0, 0>
      <<<dim3(128, 8), 256, 0, stream>>>(X32, WCB, CN, nullptr, SCOR,
                                         M32, 512, 256, -2.f, 1);
  k_argmin_zq<<<M32, 256, 0, stream>>>(SCOR, codebook, zq);

  // ---- decoder ----
  resblock(2, ze, P);
  resblock(3, P, Q);

  k_bn_stats<<<256, 256, 0, stream>>>(Q, dbn1_g, dbn1_b, SC, SH, 256, 1024, 16);
  k_bn_relu_f16<<<grid1(E32), 256, 0, stream>>>(Q, SC, SH, X32, E32, 10);
  // dct1: ConvT 256->256 k4 s2 p1, (256,32,32) -> (256,64,64): M=65536 K=4096
  k_gemm_wmma<MODE_CONVT, 4, 4, 0, 0, 5, 5, 6, 6>
      <<<dim3(512, 4), 256, 0, stream>>>(X32, WDT1, dct1_b, nullptr, A64,
                                         16 * 64 * 64, 256, 4096, 1.f, 0);
  k_bn_stats<<<256, 256, 0, stream>>>(A64, dbn2_g, dbn2_b, SC, SH, 256, 4096, 16);
  k_bn_relu_f16<<<grid1(E64), 256, 0, stream>>>(A64, SC, SH, X64, E64, 12);
  k_convT_dct2<<<grid1(16 * 3 * 128 * 128), 256, 0, stream>>>(X64, dct2_w, dct2_b, xrec);
}
